// GraphAttentionLayer_28535762714791
// MI455X (gfx1250) — matrix operand order repaired and checked by reading, not compiled, not run on hardware
//
#include <hip/hip_runtime.h>
#include <hip/hip_bf16.h>

typedef __attribute__((ext_vector_type(16))) _Float16 v16h;
typedef __attribute__((ext_vector_type(8)))  _Float16 v8h;
typedef __attribute__((ext_vector_type(4)))  _Float16 v4h;
typedef __attribute__((ext_vector_type(8)))  float    v8f;
typedef __attribute__((ext_vector_type(4)))  unsigned int v4u;
typedef __attribute__((ext_vector_type(8)))  int      v8i;
typedef __attribute__((ext_vector_type(4)))  int      v4i;

#define ALPHA_   0.2f
#define NEG_INF_ -9.0e15f
#define B_  8
#define N_  2048
#define D_  256

// TDM-staged WhT tile: 256 o-rows x 32 j (f16) = 64B/row + 8B pad -> 72B stride
#define TDM_ROW_BYTES 72
#define TDM_BUF_BYTES (D_ * TDM_ROW_BYTES)   // 18432

// ---------------------------------------------------------------------------
// Tensor Data Mover issue: load 2D tile (tile_dim0=32 j-halves, tile_dim1=256
// o-rows, row stride = N_ halves) from WhT into LDS, with LDS row padding of
// 2 DWORDs after every 16 DWORDs (pad_interval=3 -> 16dw, pad_amount=1 -> 2dw).
// D# bit-packing per CDNA5 ISA 8.3/8.4. Tracked by TENSORcnt.
// ---------------------------------------------------------------------------
__device__ __forceinline__ void tdm_load_tile(const _Float16* gbase, unsigned lds_addr) {
    unsigned long long ga = (unsigned long long)(size_t)gbase;
    v4u g0;
    g0[0] = 1u;                                   // count=1, user descriptor
    g0[1] = lds_addr;                             // lds_addr [63:32]
    g0[2] = (unsigned)ga;                         // global_addr[31:0]
    g0[3] = (unsigned)((ga >> 32) & 0x01FFFFFFu)  // global_addr[56:32]
            | (2u << 30);                         // type=2 ("image")
    v8i g1;
    g1[0] = (int)((1u << 16)                      // data_size = 2 bytes
                | (1u << 20)                      // pad_enable
                | (3u << 22)                      // pad_interval: 16 DWORDs
                | (1u << 25));                    // pad_amount: 2 DWORDs (8B)
    g1[1] = (int)(((unsigned)N_ & 0xFFFFu) << 16);        // tensor_dim0[15:0]
    g1[2] = (int)((N_ >> 16) | (((unsigned)D_ & 0xFFFFu) << 16)); // tdim0 hi | tdim1 lo
    g1[3] = (int)((D_ >> 16) | (32u << 16));              // tdim1 hi | tile_dim0=32
    g1[4] = (int)((unsigned)D_);                          // tile_dim1=256, tile_dim2=0
    g1[5] = (int)((unsigned)N_);                          // tensor_dim0_stride lo
    g1[6] = 0;                                            // stride hi | tdim1_stride lo
    g1[7] = 0;
    v4i gz; gz[0] = 0; gz[1] = 0; gz[2] = 0; gz[3] = 0;   // groups 2/3 unused (2D)
#if __has_include(<hip/amd_detail/amd_gfx1250_TDM.h>)
    v8i gz8; gz8[0]=0; gz8[1]=0; gz8[2]=0; gz8[3]=0; gz8[4]=0; gz8[5]=0; gz8[6]=0; gz8[7]=0;
    __builtin_amdgcn_tensor_load_to_lds(g0, g1, gz, gz, gz8, 0);   // 6-arg toolchain
#else
    __builtin_amdgcn_tensor_load_to_lds(g0, g1, gz, gz, 0);        // 5-arg ROCm 7.2
#endif
}

// ---------------------------------------------------------------------------
// Kernel 0: convert W (D x D f32) to f16 workspace
// ---------------------------------------------------------------------------
__global__ void k_cvt_w(const float* __restrict__ W, _Float16* __restrict__ w16) {
    int i = blockIdx.x * 256 + threadIdx.x;
    w16[i] = (_Float16)W[i];
}

// ---------------------------------------------------------------------------
// Kernel 1: Wh = h @ W^T via WMMA f16; store WhT_f16[b][o][n]; compute s1,s2.
// ---------------------------------------------------------------------------
__global__ __launch_bounds__(128)
void k_gemm1(const float* __restrict__ h,
             const _Float16* __restrict__ w16,
             const float* __restrict__ avec,
             _Float16* __restrict__ whT,
             float* __restrict__ s1,
             float* __restrict__ s2) {
    const int b  = blockIdx.y;
    const int n0 = blockIdx.x * 16;
    const int tid = threadIdx.x;
    const int wv  = tid >> 5;
    const int L   = tid & 31;
    const int m   = L & 15;
    const bool lo = (L < 16);

    __shared__ float s1sh[4][16];
    __shared__ float s2sh[4][16];

    v8f acc[4];
#pragma unroll
    for (int i = 0; i < 4; ++i)
#pragma unroll
        for (int r = 0; r < 8; ++r) acc[i][r] = 0.0f;

    const float* hrow = h + ((size_t)b * N_ + n0 + m) * D_;

#pragma unroll 1
    for (int ks = 0; ks < 8; ++ks) {
        const int kb = ks * 32;
        const float4* pA = (const float4*)(hrow + kb + (lo ? 0 : 8));
        const float4* pB = (const float4*)(hrow + kb + (lo ? 16 : 24));
        float4 a0 = pA[0], a1 = pA[1], b0 = pB[0], b1 = pB[1];
        v16h af;
        af[0]=(_Float16)a0.x; af[1]=(_Float16)a0.y; af[2]=(_Float16)a0.z; af[3]=(_Float16)a0.w;
        af[4]=(_Float16)a1.x; af[5]=(_Float16)a1.y; af[6]=(_Float16)a1.z; af[7]=(_Float16)a1.w;
        af[8]=(_Float16)b0.x; af[9]=(_Float16)b0.y; af[10]=(_Float16)b0.z; af[11]=(_Float16)b0.w;
        af[12]=(_Float16)b1.x; af[13]=(_Float16)b1.y; af[14]=(_Float16)b1.z; af[15]=(_Float16)b1.w;

#pragma unroll
        for (int obl = 0; obl < 4; ++obl) {
            const int o = (wv * 4 + obl) * 16 + m;
            const v8h* bp = (const v8h*)(w16 + (size_t)o * D_ + kb + (lo ? 0 : 8));
            v8h bl0 = bp[0], bl1 = bp[2];
            v16h bf;
#pragma unroll
            for (int t = 0; t < 8; ++t) { bf[t] = bl0[t]; bf[8 + t] = bl1[t]; }
            acc[obl] = __builtin_amdgcn_wmma_f32_16x16x32_f16(
                false, af, false, bf, (short)0, acc[obl], false, false);
        }
    }

    float p1[8], p2[8];
#pragma unroll
    for (int r = 0; r < 8; ++r) { p1[r] = 0.0f; p2[r] = 0.0f; }

#pragma unroll
    for (int obl = 0; obl < 4; ++obl) {
        const int o = (wv * 4 + obl) * 16 + m;
        const float a1v = avec[o];
        const float a2v = avec[D_ + o];
        v8h oh;
#pragma unroll
        for (int r = 0; r < 8; ++r) {
            float v = acc[obl][r];
            p1[r] += v * a1v;
            p2[r] += v * a2v;
            oh[r] = (_Float16)v;
        }
        *(v8h*)(whT + ((size_t)b * D_ + o) * N_ + n0 + (lo ? 0 : 8)) = oh;
    }

#pragma unroll
    for (int msk = 1; msk <= 8; msk <<= 1)
#pragma unroll
        for (int r = 0; r < 8; ++r) {
            p1[r] += __shfl_xor(p1[r], msk);
            p2[r] += __shfl_xor(p2[r], msk);
        }

    if (m == 0) {
        const int rb = lo ? 0 : 8;
#pragma unroll
        for (int r = 0; r < 8; ++r) { s1sh[wv][rb + r] = p1[r]; s2sh[wv][rb + r] = p2[r]; }
    }
    __syncthreads();
    if (tid < 16) {
        s1[(size_t)b * N_ + n0 + tid] = s1sh[0][tid] + s1sh[1][tid] + s1sh[2][tid] + s1sh[3][tid];
    } else if (tid < 32) {
        int r = tid - 16;
        s2[(size_t)b * N_ + n0 + r] = s2sh[0][r] + s2sh[1][r] + s2sh[2][r] + s2sh[3][r];
    }
}

// ---------------------------------------------------------------------------
// Kernel 2: fused masked-softmax attention + h' = attn @ Wh (flash style).
// 4 waves split j; per-wave WhT j-tiles staged via TDM, double buffered in
// LDS; adj/s2 stream double-buffered in registers; cross-wave LSE merge.
// ---------------------------------------------------------------------------
__global__ __launch_bounds__(128)
void k_gat(const int* __restrict__ adj,
           const _Float16* __restrict__ whT,
           const float* __restrict__ s1,
           const float* __restrict__ s2,
           float* __restrict__ out) {
    const int b  = blockIdx.y;
    const int i0 = blockIdx.x * 16;
    const int tid = threadIdx.x;
    const int wv  = tid >> 5;
    const int wvu = __builtin_amdgcn_readfirstlane(wv);   // uniform wave id
    const int L   = tid & 31;
    const int m   = L & 15;
    const bool lo = (L < 16);
    const int cbase = lo ? 0 : 8;

    __shared__ __align__(16) char tdm_buf[4][2][TDM_BUF_BYTES];   // 144 KB
    __shared__ float msh[4][16];
    __shared__ float lsh[4][16];
    __shared__ float lfin_sh[16];
    float (*accsh)[257] = (float (*)[257])&tdm_buf[0][0][0];      // aliased after loop

    v8f acc[16];
#pragma unroll
    for (int i = 0; i < 16; ++i)
#pragma unroll
        for (int r = 0; r < 8; ++r) acc[i][r] = 0.0f;

    float mrun = -3.0e38f, lrun = 0.0f;
    const float s1v = s1[(size_t)b * N_ + i0 + m];
    const int*   arow = adj + ((size_t)b * N_ + i0 + m) * N_;
    const float* s2b  = s2 + (size_t)b * N_;
    const _Float16* whTb = whT + (size_t)b * D_ * N_;   // row o has stride N_

    // prologue: TDM tile 0, register-load adj/s2 tile 0
    tdm_load_tile(whTb + wvu * 512, (unsigned)(size_t)&tdm_buf[wvu][0][0]);

    float4 q0, q1, q2, q3; int4 c0, c1, c2, c3;
    {
        const int jA = wv * 512 + (lo ? 0 : 8);
        const int jB = wv * 512 + (lo ? 16 : 24);
        q0 = ((const float4*)(s2b + jA))[0]; q1 = ((const float4*)(s2b + jA))[1];
        q2 = ((const float4*)(s2b + jB))[0]; q3 = ((const float4*)(s2b + jB))[1];
        c0 = ((const int4*)(arow + jA))[0];  c1 = ((const int4*)(arow + jA))[1];
        c2 = ((const int4*)(arow + jB))[0];  c3 = ((const int4*)(arow + jB))[1];
    }

#pragma unroll 1
    for (int t = 0; t < 16; ++t) {
        const int j0 = wv * 512 + t * 32;

        // issue next TDM tile + next adj/s2 register tile, then wait for current
        float4 nq0{}, nq1{}, nq2{}, nq3{}; int4 nc0{}, nc1{}, nc2{}, nc3{};
        if (t < 15) {
            tdm_load_tile(whTb + wvu * 512 + (t + 1) * 32,
                          (unsigned)(size_t)&tdm_buf[wvu][(t + 1) & 1][0]);
            const int jA = j0 + 32 + (lo ? 0 : 8);
            const int jB = j0 + 32 + (lo ? 16 : 24);
            nq0 = ((const float4*)(s2b + jA))[0]; nq1 = ((const float4*)(s2b + jA))[1];
            nq2 = ((const float4*)(s2b + jB))[0]; nq3 = ((const float4*)(s2b + jB))[1];
            nc0 = ((const int4*)(arow + jA))[0];  nc1 = ((const int4*)(arow + jA))[1];
            nc2 = ((const int4*)(arow + jB))[0];  nc3 = ((const int4*)(arow + jB))[1];
        }

        float sv[16] = { q0.x,q0.y,q0.z,q0.w, q1.x,q1.y,q1.z,q1.w,
                         q2.x,q2.y,q2.z,q2.w, q3.x,q3.y,q3.z,q3.w };
        int   av[16] = { c0.x,c0.y,c0.z,c0.w, c1.x,c1.y,c1.z,c1.w,
                         c2.x,c2.y,c2.z,c2.w, c3.x,c3.y,c3.z,c3.w };
        float e[16];
#pragma unroll
        for (int k = 0; k < 16; ++k) {
            float ev = s1v + sv[k];
            ev = (ev > 0.0f) ? ev : ALPHA_ * ev;
            e[k] = (av[k] > 0) ? ev : NEG_INF_;
        }
        float mx = e[0];
#pragma unroll
        for (int k = 1; k < 16; ++k) mx = fmaxf(mx, e[k]);
        mx = fmaxf(mx, __shfl_xor(mx, 16));
        const float mnew  = fmaxf(mrun, mx);
        const float scale = __expf(mrun - mnew);

        float p[16], ls = 0.0f;
#pragma unroll
        for (int k = 0; k < 16; ++k) { p[k] = __expf(e[k] - mnew); ls += p[k]; }
        ls += __shfl_xor(ls, 16);
        lrun = lrun * scale + ls;
        mrun = mnew;

        float sc[8];
#pragma unroll
        for (int r = 0; r < 8; ++r) sc[r] = __shfl(scale, cbase + r);
#pragma unroll
        for (int ob = 0; ob < 16; ++ob)
#pragma unroll
            for (int r = 0; r < 8; ++r) acc[ob][r] *= sc[r];

        v16h af;
#pragma unroll
        for (int k = 0; k < 16; ++k) af[k] = (_Float16)p[k];

        // current tile's TDM transfer must be complete (<=1 outstanding left)
        if (t < 15) __builtin_amdgcn_s_wait_tensorcnt(1);
        else        __builtin_amdgcn_s_wait_tensorcnt(0);

        const char* buf = &tdm_buf[wv][t & 1][0];
#pragma unroll
        for (int ob = 0; ob < 16; ++ob) {
            const int o = ob * 16 + m;
            const v4h* bp = (const v4h*)(buf + (size_t)o * TDM_ROW_BYTES + (lo ? 0 : 16));
            v4h r0 = bp[0], r1 = bp[1], r2 = bp[4], r3 = bp[5];   // ds_load_b64 x4
            v16h bf;
#pragma unroll
            for (int tt = 0; tt < 4; ++tt) {
                bf[tt] = r0[tt]; bf[4 + tt] = r1[tt]; bf[8 + tt] = r2[tt]; bf[12 + tt] = r3[tt];
            }
            acc[ob] = __builtin_amdgcn_wmma_f32_16x16x32_f16(
                false, af, false, bf, (short)0, acc[ob], false, false);
        }

        if (t < 15) { q0=nq0; q1=nq1; q2=nq2; q3=nq3; c0=nc0; c1=nc1; c2=nc2; c3=nc3; }
    }

    // ---- cross-wave merge (accsh aliases the TDM buffers: barrier first) ----
    __syncthreads();
    for (int idx = tid; idx < 16 * 257; idx += 128) ((float*)accsh)[idx] = 0.0f;
    if (lo) { msh[wv][m] = mrun; lsh[wv][m] = lrun; }
    __syncthreads();

    float mf = msh[0][m];
#pragma unroll
    for (int w = 1; w < 4; ++w) mf = fmaxf(mf, msh[w][m]);
    float lf = 0.0f;
#pragma unroll
    for (int w = 0; w < 4; ++w) lf += lsh[w][m] * __expf(msh[w][m] - mf);
    if (wv == 0 && lo) lfin_sh[m] = lf;

    const float fac = __expf(mrun - mf);
    float fc[8];
#pragma unroll
    for (int r = 0; r < 8; ++r) fc[r] = __shfl(fac, cbase + r);

#pragma unroll
    for (int ob = 0; ob < 16; ++ob)
#pragma unroll
        for (int r = 0; r < 8; ++r)
            atomicAdd(&accsh[cbase + r][ob * 16 + m], acc[ob][r] * fc[r]);  // ds_add_f32
    __syncthreads();

    float* orow = out + ((size_t)b * N_ + i0) * D_;
    for (int idx = tid; idx < 16 * D_; idx += 128) {
        const int r = idx >> 8;
        const int o = idx & 255;
        orow[(size_t)r * D_ + o] = accsh[r][o] / lfin_sh[r];
    }
}

// ---------------------------------------------------------------------------
extern "C" void kernel_launch(void* const* d_in, const int* in_sizes, int n_in,
                              void* d_out, int out_size, void* d_ws, size_t ws_size,
                              hipStream_t stream) {
    (void)in_sizes; (void)n_in; (void)out_size; (void)ws_size;
    const float* h   = (const float*)d_in[0];
    const int*   adj = (const int*)d_in[1];
    const float* W   = (const float*)d_in[2];
    const float* a   = (const float*)d_in[3];
    float* out = (float*)d_out;

    char* ws = (char*)d_ws;
    _Float16* w16 = (_Float16*)ws;
    _Float16* whT = (_Float16*)(ws + (size_t)(128 << 10));
    float* s1 = (float*)(ws + (size_t)(128 << 10) + ((size_t)8 << 20));
    float* s2 = s1 + (size_t)B_ * N_;

    k_cvt_w<<<dim3((D_ * D_) / 256), dim3(256), 0, stream>>>(W, w16);
    k_gemm1<<<dim3(N_ / 16, B_), dim3(128), 0, stream>>>(h, w16, a, whT, s1, s2);
    k_gat  <<<dim3(N_ / 16, B_), dim3(128), 0, stream>>>(adj, whT, s1, s2, out);
}
